// PAM_24541443129855
// MI455X (gfx1250) — matrix operand order repaired and checked by reading, compiled, not run on hardware
//
#include <hip/hip_runtime.h>

typedef __attribute__((ext_vector_type(16))) _Float16 v16h;
typedef __attribute__((ext_vector_type(8)))  float    v8f;
typedef __attribute__((ext_vector_type(2)))  float    v2f;

#define BATCH 2
#define HW    9216          // 96*96
#define CCH   50            // channels
#define CPAD  64            // padded channels for WMMA N tiles
#define DPAD  8             // padded projection dim (6 -> 8)
#define KT    32            // key tile (WMMA K for f16)
#define NTIL  (HW / KT)     // 288 key tiles
#define QBLK  16            // query rows per wave (WMMA M)

// ---------------------------------------------------------------------------
// Kernel 1: projections. vb/vc f32 padded to 8 (zero pad doubles as the K=6,7
// padding of the f32 WMMA contraction). vd f16 padded to 64 channels, stored
// so each lane's 8 B-matrix dwords are CONTIGUOUS:
//   u32 index = (((b*NTIL + tile)*4 + nb)*32 + (c%16) + (kin&16))*8 + (kin&15)/2
//   low u16 = K even, high u16 = K odd
// ---------------------------------------------------------------------------
__global__ __launch_bounds__(256)
void pam_proj(const float* __restrict__ x,  const float* __restrict__ Wb,
              const float* __restrict__ Wc, const float* __restrict__ Wd,
              float* __restrict__ vb, float* __restrict__ vc,
              _Float16* __restrict__ vd_sw) {
  int t = blockIdx.x * blockDim.x + threadIdx.x;
  if (t >= BATCH * HW) return;
  int b = t / HW, k = t - b * HW;

  const float* xr = x + (size_t)t * CCH;
  float xv[CCH];
#pragma unroll
  for (int d = 0; d < CCH; ++d) xv[d] = xr[d];

  // vb, vc : hw x 6 (padded to 8 with zeros)
#pragma unroll
  for (int c = 0; c < 6; ++c) {
    float sb = 0.f, sc = 0.f;
    for (int d = 0; d < CCH; ++d) {
      sb = fmaf(xv[d], Wb[d * 6 + c], sb);
      sc = fmaf(xv[d], Wc[d * 6 + c], sc);
    }
    vb[(size_t)t * DPAD + c] = sb;
    vc[(size_t)t * DPAD + c] = sc;
  }
  vb[(size_t)t * DPAD + 6] = 0.f; vb[(size_t)t * DPAD + 7] = 0.f;
  vc[(size_t)t * DPAD + 6] = 0.f; vc[(size_t)t * DPAD + 7] = 0.f;

  // vd : hw x 50 -> f16, padded to 64, B-matrix swizzled (lane-contiguous)
  int tile = k / KT, kin = k - tile * KT;
  int halfofs = kin & 16;          // lanes 16-31 hold K=16..31
  int vreg    = (kin & 15) >> 1;   // dword index within the lane's 8
  int slot    = kin & 1;           // low/high 16 bits of the packed dword
  size_t tbase = (size_t)(b * NTIL + tile) * 4;
  for (int c = 0; c < CPAD; ++c) {
    float s = 0.f;
    if (c < CCH)
      for (int d = 0; d < CCH; ++d) s = fmaf(xv[d], Wd[d * CCH + c], s);
    int nb = c >> 4, cl = c & 15;
    size_t idx32 = ((tbase + nb) * 32 + cl + halfofs) * 8 + vreg;
    vd_sw[idx32 * 2 + slot] = (_Float16)s;
  }
}

// ---------------------------------------------------------------------------
// Kernel 2: per-query row max of the score matrix, on the matrix pipe.
// Same score WMMAs as pam_attn (bitwise-identical), so exp(s-m) <= 1 exactly.
// Per-lane partial maxima live in C/D layout; one cross-lane max at the end.
// ---------------------------------------------------------------------------
__global__ __launch_bounds__(32)
void pam_rowmax(const float* __restrict__ vb, const float* __restrict__ vc,
                float* __restrict__ mrow) {
  const int lane = threadIdx.x;
  const int blk  = blockIdx.x;
  const int b    = blk / (HW / QBLK);
  const int q0   = (blk - b * (HW / QBLK)) * QBLK;
  const int hi8  = (lane >> 4) << 3;

  const size_t qidx = (size_t)b * HW + q0 + (lane & 15);
  const int kb = (lane >> 4) * 2;
  const float* vbrow = vb + qidx * DPAD;
  v2f A0, A1;
  A0.x = vbrow[kb];     A0.y = vbrow[kb + 1];
  A1.x = vbrow[4 + kb]; A1.y = vbrow[4 + kb + 1];

  const float* vcb = vc + (size_t)b * HW * DPAD;
  v8f M;
#pragma unroll
  for (int i = 0; i < 8; ++i) M[i] = -3.4e38f;

  for (int tile = 0; tile < NTIL; ++tile) {
#pragma unroll
    for (int sub = 0; sub < 2; ++sub) {
      int key = tile * KT + sub * 16 + (lane & 15);
      const float* vcrow = vcb + (size_t)key * DPAD;
      v2f B0, B1;
      B0.x = vcrow[kb];     B0.y = vcrow[kb + 1];
      B1.x = vcrow[4 + kb]; B1.y = vcrow[4 + kb + 1];
      v8f S = {};
      S = __builtin_amdgcn_wmma_f32_16x16x4_f32(
          false, A0, false, B0, (short)0, S, false, false);
      S = __builtin_amdgcn_wmma_f32_16x16x4_f32(
          false, A1, false, B1, (short)0, S, false, false);
#pragma unroll
      for (int i = 0; i < 8; ++i) M[i] = fmaxf(M[i], S[i]);
    }
  }
  // cross-lane max within each 16-lane half (masks 1,2,4,8 stay in-half)
#pragma unroll
  for (int d = 1; d < 16; d <<= 1)
#pragma unroll
    for (int i = 0; i < 8; ++i) M[i] = fmaxf(M[i], __shfl_xor(M[i], d));
#pragma unroll
  for (int i = 0; i < 8; ++i)
    if ((lane & 15) == i) mrow[(size_t)b * HW + q0 + hi8 + i] = M[i];
}

// ---------------------------------------------------------------------------
// Kernel 3: flash-style attention, fully on the matrix pipe.
// Per 32-key tile: 4x v_wmma_f32_16x16x4_f32 compute scores (row max folded
// into C operand), exp -> f16 -> LDS transpose, row-sum partials accumulated
// in-register, then 4x v_wmma_f32_16x16x32_f16 accumulate 64 channels.
// The 1/rowsum normalization is folded into the epilogue scale with gamma.
// ---------------------------------------------------------------------------
__global__ __launch_bounds__(32)
void pam_attn(const float* __restrict__ x,  const float* __restrict__ vb,
              const float* __restrict__ vc, const _Float16* __restrict__ vd_sw,
              const float* __restrict__ mrow, const float* __restrict__ gamma,
              float* __restrict__ out) {
  __shared__ _Float16 p_lds[QBLK * KT];   // 16 x 32 f16 probability tile

  const int lane = threadIdx.x;
  const int blk  = blockIdx.x;
  const int b    = blk / (HW / QBLK);
  const int q0   = (blk - b * (HW / QBLK)) * QBLK;
  const int hi8  = (lane >> 4) << 3;      // 0 for lanes 0-15, 8 for 16-31

  // f32-WMMA A operand (16x4): lane = M row, 2 consecutive K per lane.
  const size_t qidx = (size_t)b * HW + q0 + (lane & 15);
  const int kb = (lane >> 4) * 2;         // K base: 0 (lo half) / 2 (hi half)
  const float* vbrow = vb + qidx * DPAD;
  v2f A0, A1;
  A0.x = vbrow[kb];     A0.y = vbrow[kb + 1];      // K = kb, kb+1
  A1.x = vbrow[4 + kb]; A1.y = vbrow[4 + kb + 1];  // K = 4+kb, 5+kb (6,7 = 0 pad)

  // -rowmax folded into the score accumulator: p' = exp(s - m)
  float offlane = -mrow[qidx];
  v8f offc;
#pragma unroll
  for (int i = 0; i < 8; ++i) offc[i] = __shfl(offlane, i + hi8);

  const float*    vcb = vc + (size_t)b * HW * DPAD;
  const unsigned* vdb = (const unsigned*)vd_sw + (size_t)b * NTIL * 4 * 8 * 32;
  const unsigned* plu = (const unsigned*)p_lds;

  v8f acc[4] = {};   // 16 x 64 f32 accumulator (C/D layout, 4 N-blocks)
  v8f lsum   = {};   // per-lane partial row sums of p' (C/D layout)

  for (int tile = 0; tile < NTIL; ++tile) {
    // ---- scores + softmax numerator for two 16-key subtiles ----
#pragma unroll
    for (int sub = 0; sub < 2; ++sub) {
      int key = tile * KT + sub * 16 + (lane & 15);
      const float* vcrow = vcb + (size_t)key * DPAD;
      v2f B0, B1;                                   // B (4x16): N=lane, 2 K/lane
      B0.x = vcrow[kb];     B0.y = vcrow[kb + 1];
      B1.x = vcrow[4 + kb]; B1.y = vcrow[4 + kb + 1];
      v8f S = __builtin_amdgcn_wmma_f32_16x16x4_f32(
          false, A0, false, B0, (short)0, offc, false, false);
      S = __builtin_amdgcn_wmma_f32_16x16x4_f32(
          false, A1, false, B1, (short)0, S, false, false);
#pragma unroll
      for (int i = 0; i < 8; ++i) {
        float P = __expf(S[i]);                     // <= 1 by construction
        lsum[i] += P;
        p_lds[(hi8 + i) * KT + sub * 16 + (lane & 15)] = (_Float16)P;
      }
    }
    __syncthreads();

    // ---- A-matrix (16x32 f16, ISA 7.12.2) from LDS: 2x ds_load_b128 ----
    int abase = (lane & 15) * (KT / 2) + ((lane >> 4) << 3);
    union { unsigned u[8]; v16h h; } A;
#pragma unroll
    for (int v = 0; v < 4; ++v) {
      A.u[v]     = plu[abase + v];
      A.u[4 + v] = plu[abase + 4 + v];
    }

    // ---- B (vd) loads: 8 contiguous dwords per lane = 2x b128 per nb ----
    const uint4* bt = (const uint4*)(vdb + ((size_t)tile * 4) * 256 + lane * 8);
#pragma unroll
    for (int nb = 0; nb < 4; ++nb) {
      union { uint4 q[2]; v16h h; } Bm;
      Bm.q[0] = bt[nb * 64];           // (nb*256 + lane*8) dwords
      Bm.q[1] = bt[nb * 64 + 1];
      acc[nb] = __builtin_amdgcn_wmma_f32_16x16x32_f16(
          false, A.h, false, Bm.h, (short)0, acc[nb], false, false);
    }
    __syncthreads();   // protect p_lds WAR for next tile
  }

  // ---- finish row sums: reduce across each 16-lane half ----
#pragma unroll
  for (int d = 1; d < 16; d <<= 1)
#pragma unroll
    for (int i = 0; i < 8; ++i) lsum[i] += __shfl_xor(lsum[i], d);

  // Epilogue: out = (gamma / rowsum) * acc + x.
  const float g = gamma[0];
#pragma unroll
  for (int i = 0; i < 8; ++i) {
    float gi = g / lsum[i];
    size_t obase = ((size_t)b * HW + q0 + hi8 + i) * CCH;
#pragma unroll
    for (int nb = 0; nb < 4; ++nb) {
      int c = nb * 16 + (lane & 15);
      if (c < CCH) out[obase + c] = fmaf(gi, acc[nb][i], x[obase + c]);
    }
  }
}

// ---------------------------------------------------------------------------
extern "C" void kernel_launch(void* const* d_in, const int* in_sizes, int n_in,
                              void* d_out, int out_size, void* d_ws, size_t ws_size,
                              hipStream_t stream) {
  const float* x     = (const float*)d_in[0];
  const float* Wb    = (const float*)d_in[1];
  const float* Wc    = (const float*)d_in[2];
  const float* Wd    = (const float*)d_in[3];
  const float* gamma = (const float*)d_in[4];
  float* out = (float*)d_out;

  // Workspace layout (~3.6 MB total; fully L2-resident on MI455X's 192 MB L2)
  constexpr size_t VB_B = (size_t)BATCH * HW * DPAD * sizeof(float);   // 589824
  constexpr size_t VD_B = (size_t)BATCH * HW * CPAD * sizeof(_Float16);// 2359296
  char* ws = (char*)d_ws;
  float*    vb   = (float*)(ws);
  float*    vc   = (float*)(ws + VB_B);
  _Float16* vd   = (_Float16*)(ws + 2 * VB_B);
  float*    mrow = (float*)(ws + 2 * VB_B + VD_B);
  (void)ws_size; (void)in_sizes; (void)n_in; (void)out_size;

  const int nrows  = BATCH * HW;                  // 18432
  const int nwaves = BATCH * (HW / QBLK);         // 1152
  pam_proj  <<<(nrows + 255) / 256, 256, 0, stream>>>(x, Wb, Wc, Wd, vb, vc, vd);
  pam_rowmax<<<nwaves, 32, 0, stream>>>(vb, vc, mrow);
  pam_attn  <<<nwaves, 32, 0, stream>>>(x, vb, vc, vd, mrow, gamma, out);
}
